// TemporalAttentionPooling_86955907874973
// MI455X (gfx1250) — matrix operand order repaired and checked by reading, not compiled, not run on hardware
//
#include <hip/hip_runtime.h>

// ---------------------------------------------------------------------------
// TemporalAttentionPooling for MI455X (gfx1250, wave32)
//
// Roofline: x is 308MB fp32; algorithm requires 2 full reads of x (squeeze
// reduction, then attended+multiscale pass) + 19MB write => ~635MB traffic,
// ~27us at 23.3 TB/s. GEMMs are 512x192x48 (~19 MFLOP) => negligible; they
// are mapped onto v_wmma_f32_16x16x32_f16 (one 16x16 tile per wave) so the
// matrix work uses the CDNA5 matrix pipe.
// ---------------------------------------------------------------------------

#define C_DIM  192
#define CR_DIM 48
#define B_DIM  32
#define T_DIM  16
#define HW     784      // 28*28
#define HW4    196      // HW / 4 (float4 chunks)
#define M_DIM  512      // B*T rows

typedef __attribute__((ext_vector_type(16))) _Float16 v16h;
typedef __attribute__((ext_vector_type(8)))  float    v8f;

// ---------------------------------------------------------------------------
// Kernel 1: squeezed[m][c] = mean over H*W of x[b,c,t,:,:]   (f32 -> f16)
// One block per (b,c,t); x is contiguous per (b,c,t) row (784 floats).
// ---------------------------------------------------------------------------
__global__ void squeeze_kernel(const float* __restrict__ x,
                               _Float16* __restrict__ S) {
    __shared__ float sm[128];
    const int bct = blockIdx.x;                 // ((b*192 + c)*16 + t)
    const float* row = x + (size_t)bct * HW;
    float s = 0.0f;
    for (int i = threadIdx.x; i < HW; i += 128) s += row[i];
    sm[threadIdx.x] = s;
    __syncthreads();
    for (int off = 64; off > 0; off >>= 1) {
        if ((int)threadIdx.x < off) sm[threadIdx.x] += sm[threadIdx.x + off];
        __syncthreads();
    }
    if (threadIdx.x == 0) {
        const int t  = bct & 15;
        const int bc = bct >> 4;
        const int c  = bc % C_DIM;
        const int b  = bc / C_DIM;
        const int m  = b * T_DIM + t;           // row index in (B*T, C)
        S[m * C_DIM + c] = (_Float16)(sm[0] * (1.0f / (float)HW));
    }
}

// ---------------------------------------------------------------------------
// Kernel 2: H[m][o] = relu( sum_c S[m][c] * w1[o][c] + b1[o] )
// M=512, N=48, K=192. One 16x16 tile per wave, 6 K-steps of 32.
// Fragment layouts per CDNA5 ISA 7.12.2 (wave32).
// ---------------------------------------------------------------------------
__global__ void gemm1_kernel(const _Float16* __restrict__ S,
                             const float* __restrict__ w1,
                             const float* __restrict__ b1,
                             _Float16* __restrict__ H) {
    const int wave = blockIdx.x * (blockDim.x >> 5) + (threadIdx.x >> 5);
    const int lane = threadIdx.x & 31;
    const int half = lane >> 4;                 // 0: lanes 0-15, 1: lanes 16-31
    const int l16  = lane & 15;
    const int mT = wave / 3, nT = wave % 3;     // 32 x 3 tiles
    const int mBase = mT * 16, nBase = nT * 16;

    v8f acc = {};
    for (int kt = 0; kt < 6; ++kt) {
        v16h a, b;
        #pragma unroll
        for (int i = 0; i < 16; ++i) {
            // A (16x32 f16): lane half selects K+8; idx 0-7 -> K 0-7, 8-15 -> K 16-23
            const int ka = kt * 32 + ((i & 7) | ((i >> 3) << 4) | (half << 3));
            a[i] = S[(mBase + l16) * C_DIM + ka];
            // B (32x16 f16): lane = N; same K order as A (half-wave selects K+8)
            const int kb = kt * 32 + ((i & 7) | ((i >> 3) << 4) | (half << 3));
            b[i] = (_Float16)w1[(nBase + l16) * C_DIM + kb];
        }
        acc = __builtin_amdgcn_wmma_f32_16x16x32_f16(
            /*neg_a=*/false, a, /*neg_b=*/false, b,
            /*c_mod=*/(short)0, acc, /*reuse_a=*/false, /*reuse_b=*/false);
    }
    const float bias = b1[nBase + l16];
    #pragma unroll
    for (int r = 0; r < 8; ++r) {
        // C/D (16x16 f32): VGPR r holds M = r + 8*half, N = lane&15
        const int m = mBase + r + (half << 3);
        float v = acc[r] + bias;
        H[m * CR_DIM + (nBase + l16)] = (_Float16)(v > 0.0f ? v : 0.0f);
    }
}

// ---------------------------------------------------------------------------
// Kernel 3: attn[m][c] = sigmoid( sum_o H[m][o] * w2[c][o] + b2[c] )
// M=512, N=192, K=48 (2 K-steps of 32, second half zero-masked).
// ---------------------------------------------------------------------------
__global__ void gemm2_kernel(const _Float16* __restrict__ H,
                             const float* __restrict__ w2,
                             const float* __restrict__ b2,
                             float* __restrict__ attn) {
    const int wave = blockIdx.x * (blockDim.x >> 5) + (threadIdx.x >> 5);
    const int lane = threadIdx.x & 31;
    const int half = lane >> 4;
    const int l16  = lane & 15;
    const int mT = wave / 12, nT = wave % 12;   // 32 x 12 tiles
    const int mBase = mT * 16, nBase = nT * 16;

    v8f acc = {};
    for (int kt = 0; kt < 2; ++kt) {
        v16h a, b;
        #pragma unroll
        for (int i = 0; i < 16; ++i) {
            const int ka = kt * 32 + ((i & 7) | ((i >> 3) << 4) | (half << 3));
            a[i] = (ka < CR_DIM) ? H[(mBase + l16) * CR_DIM + ka] : (_Float16)0.0f;
            const int kb = kt * 32 + ((i & 7) | ((i >> 3) << 4) | (half << 3));
            b[i] = (kb < CR_DIM) ? (_Float16)w2[(nBase + l16) * CR_DIM + kb]
                                 : (_Float16)0.0f;
        }
        acc = __builtin_amdgcn_wmma_f32_16x16x32_f16(
            false, a, false, b, (short)0, acc, false, false);
    }
    const float bias = b2[nBase + l16];
    #pragma unroll
    for (int r = 0; r < 8; ++r) {
        const int m = mBase + r + (half << 3);
        const float v = acc[r] + bias;
        attn[m * C_DIM + (nBase + l16)] = 1.0f / (1.0f + __expf(-v));
    }
}

// ---------------------------------------------------------------------------
// Kernel 4: fused second pass over x (the other half of the HBM traffic).
// out[b,c,hw] = 0.5 * sum_t x[b,c,t,hw]*attn[b,t,c]
//             + 0.5 * ( w0*x[..,15,..] + (w1/8)*sum_{t>=8} x + (w2/16)*sum_t x )
// One block per (b,c); 196 threads x float4 covers the 784 spatial elements.
// ---------------------------------------------------------------------------
__global__ void fuse_kernel(const float* __restrict__ x,
                            const float* __restrict__ attn,
                            const float* __restrict__ scale_w,
                            float* __restrict__ out) {
    const int bc  = blockIdx.x;                 // b*192 + c
    const int b   = bc / C_DIM;
    const int c   = bc % C_DIM;
    const int tid = threadIdx.x;                // 0..195

    // softmax over the 3 scale weights (trivial, per-thread)
    const float e0 = __expf(scale_w[0]);
    const float e1 = __expf(scale_w[1]);
    const float e2 = __expf(scale_w[2]);
    const float inv = 1.0f / (e0 + e1 + e2);
    const float w0 = e0 * inv;
    const float w1 = e1 * inv * (1.0f / 8.0f);   // folds the mid-mean divide
    const float w2 = e2 * inv * (1.0f / 16.0f);  // folds the overall-mean divide

    float at[T_DIM];
    #pragma unroll
    for (int t = 0; t < T_DIM; ++t) at[t] = attn[(b * T_DIM + t) * C_DIM + c];

    const float4* xv = (const float4*)x;
    const size_t base = (size_t)bc * T_DIM * HW4;

    float ax = 0, ay = 0, az = 0, aw = 0;       // attended accumulator
    float ox = 0, oy = 0, oz = 0, ow = 0;       // overall sum
    float mx = 0, my = 0, mz = 0, mw = 0;       // mid (t >= 8) sum
    float rx = 0, ry = 0, rz = 0, rw = 0;       // recent (t == 15)

    #pragma unroll
    for (int t = 0; t < T_DIM; ++t) {
        if (t + 1 < T_DIM)                      // gfx1250 global_prefetch_b8
            __builtin_prefetch(&xv[base + (size_t)(t + 1) * HW4 + tid], 0, 0);
        const float4 v = xv[base + (size_t)t * HW4 + tid];
        const float a = at[t];
        ax += v.x * a; ay += v.y * a; az += v.z * a; aw += v.w * a;
        ox += v.x;     oy += v.y;     oz += v.z;     ow += v.w;
        if (t >= T_DIM / 2) { mx += v.x; my += v.y; mz += v.z; mw += v.w; }
        if (t == T_DIM - 1) { rx = v.x; ry = v.y; rz = v.z; rw = v.w; }
    }

    float4 o;
    o.x = 0.5f * ax + 0.5f * (w0 * rx + w1 * mx + w2 * ox);
    o.y = 0.5f * ay + 0.5f * (w0 * ry + w1 * my + w2 * oy);
    o.z = 0.5f * az + 0.5f * (w0 * rz + w1 * mz + w2 * oz);
    o.w = 0.5f * aw + 0.5f * (w0 * rw + w1 * mw + w2 * ow);
    ((float4*)out)[(size_t)bc * HW4 + tid] = o;
}

// ---------------------------------------------------------------------------
// Launch. Workspace layout (all < 1MB, L2-resident between kernels):
//   S    : f16 [512][192]  at offset 0        (196608 B)
//   H    : f16 [512][48]   at offset 196608   ( 49152 B)
//   attn : f32 [512][192]  at offset 245760   (393216 B)
// ---------------------------------------------------------------------------
extern "C" void kernel_launch(void* const* d_in, const int* in_sizes, int n_in,
                              void* d_out, int out_size, void* d_ws, size_t ws_size,
                              hipStream_t stream) {
    const float* x   = (const float*)d_in[0];
    const float* w1  = (const float*)d_in[1];
    const float* b1  = (const float*)d_in[2];
    const float* w2  = (const float*)d_in[3];
    const float* b2  = (const float*)d_in[4];
    const float* swt = (const float*)d_in[5];
    float* out = (float*)d_out;

    char* ws = (char*)d_ws;
    _Float16* S    = (_Float16*)(ws);
    _Float16* Hbuf = (_Float16*)(ws + (size_t)M_DIM * C_DIM * 2);
    float*    attn = (float*)(ws + (size_t)M_DIM * C_DIM * 2
                                 + (size_t)M_DIM * CR_DIM * 2);

    // Pass 1: squeeze (reads all of x, HBM-bound)
    squeeze_kernel<<<B_DIM * C_DIM * T_DIM, 128, 0, stream>>>(x, S);

    // GEMM1: 96 tiles, 1 tile/wave -> 12 blocks x 8 waves
    gemm1_kernel<<<12, 256, 0, stream>>>(S, w1, b1, Hbuf);

    // GEMM2: 384 tiles, 1 tile/wave -> 48 blocks x 8 waves
    gemm2_kernel<<<48, 256, 0, stream>>>(Hbuf, w2, b2, attn);

    // Pass 2: fused attended + multi-scale (reads all of x again, writes out)
    fuse_kernel<<<B_DIM * C_DIM, HW4, 0, stream>>>(x, attn, swt, out);
}
